// LSTMOneView_6717328851369
// MI455X (gfx1250) — compile-verified, matrix operand order checked by reading, not run on hardware
//
#include <hip/hip_runtime.h>
#include <math.h>

typedef __attribute__((ext_vector_type(16))) _Float16 v16h;
typedef __attribute__((ext_vector_type(8)))  _Float16 v8h;
typedef __attribute__((ext_vector_type(8)))  float    v8f;

// K-offset inside a 16x32 (f16) WMMA A/B fragment for element j of lane (laneHi = lane>=16)
__device__ __forceinline__ int frag_k(int j, int laneHi) {
    return (j & 7) + ((j & 8) << 1) + (laneHi << 3);
}
// Slot of value (row m_in_tile = b, k = f) inside one M-tile block of packed A fragments.
// Block layout: [kt][lane][j] with KT = 512/32 = 16  ->  block size 16*32*16 = 8192 halfs.
__device__ __forceinline__ int packA_slot(int b, int f) {
    int kt = f >> 5, r = f & 31;
    int laneHi = (r >> 3) & 1;
    int j = (r & 7) | ((r & 16) ? 8 : 0);
    int lane = b + (laneHi << 4);
    return (kt * 32 + lane) * 16 + j;
}

// ---------------- conv0: 3->32, 112->56, direct fp32, ReLU, store f16 NHWC -------------
__global__ void conv0_kernel(const float* __restrict__ x, const float* __restrict__ w,
                             const float* __restrict__ bias, _Float16* __restrict__ out, int total) {
    int idx = blockIdx.x * blockDim.x + threadIdx.x;   // ((img*56+oy)*56+ox)*32+co
    if (idx >= total) return;
    int co = idx & 31; int t1 = idx >> 5;
    int ox = t1 % 56;  int t2 = t1 / 56;
    int oy = t2 % 56;  int img = t2 / 56;
    float acc = bias[co];
    int iy0 = oy * 2 - 1, ix0 = ox * 2 - 1;
    const float* xb = x + (long)img * 3 * 112 * 112;
    for (int ci = 0; ci < 3; ++ci)
        for (int kh = 0; kh < 3; ++kh) {
            int iy = iy0 + kh; if ((unsigned)iy >= 112u) continue;
            for (int kw = 0; kw < 3; ++kw) {
                int ix = ix0 + kw; if ((unsigned)ix >= 112u) continue;
                acc += xb[(ci * 112 + iy) * 112 + ix] * w[((co * 3 + ci) * 3 + kh) * 3 + kw];
            }
        }
    out[idx] = (_Float16)fmaxf(acc, 0.f);
}

// ------- weight repack: conv OIHW -> WMMA B fragments, K = (kh*3+kw)*Cin + ci ----------
__global__ void pack_convw_kernel(const float* __restrict__ w, _Float16* __restrict__ bp,
                                  int Cin, int cinShift, int Cout) {
    int K = Cin * 9, KT = K >> 5;
    int total = (Cout >> 4) * KT * 512;
    int e = blockIdx.x * blockDim.x + threadIdx.x;
    if (e >= total) return;
    int j = e & 15; int lane = (e >> 4) & 31; int kt = (e >> 9) % KT; int nt = e / (KT * 512);
    int k = kt * 32 + frag_k(j, lane >> 4);
    int n = nt * 16 + (lane & 15);
    int ci = k & (Cin - 1); int s = k >> cinShift;
    int kh = s / 3, kw = s - kh * 3;
    bp[e] = (_Float16)w[((n * Cin + ci) * 3 + kh) * 3 + kw];
}

// ---------------- weight repack: wx/wh [K=512][N=2048] -> WMMA B fragments -------------
__global__ void pack_w_kernel(const float* __restrict__ w, _Float16* __restrict__ bp) {
    int e = blockIdx.x * blockDim.x + threadIdx.x;
    if (e >= 128 * 16 * 512) return;
    int j = e & 15; int lane = (e >> 4) & 31; int kt = (e >> 9) & 15; int nt = e >> 13;
    int k = kt * 32 + frag_k(j, lane >> 4);
    int n = nt * 16 + (lane & 15);
    bp[e] = (_Float16)w[k * 2048 + n];
}

// ------- implicit-GEMM conv via WMMA (stride2 pad1 3x3, ReLU), NHWC activations --------
// K-order: k = (kh*3+kw)*Cin + ci  ->  per kt: one tap position, 32 contiguous channels.
// A-fragment = two contiguous 8-channel runs per lane -> two b128 loads.
__global__ void conv_wmma_kernel(const _Float16* __restrict__ in, const _Float16* __restrict__ bp,
                                 const float* __restrict__ bias, void* __restrict__ out,
                                 int Cin, int c2Shift, int Hin, int Win,
                                 int Cout, int Hout, int Wout, int numImg, int out_fp16) {
    int KT = (Cin * 9) >> 5, NT = Cout >> 4;
    int HW = Hout * Wout;
    long M = (long)numImg * HW;
    long numTiles = (M >> 4) * NT;
    int lane = threadIdx.x & 31, laneHi = lane >> 4;
    long wave  = (long)blockIdx.x * (blockDim.x >> 5) + (threadIdx.x >> 5);
    long nWave = (long)gridDim.x * (blockDim.x >> 5);
    for (long tile = wave; tile < numTiles; tile += nWave) {
        long mt = tile / NT; int nt = (int)(tile % NT);
        long m = mt * 16 + (lane & 15);
        int img = (int)(m / HW); int rem = (int)(m % HW);
        int oy = rem / Wout, ox = rem - oy * Wout;
        int iy0 = oy * 2 - 1, ix0 = ox * 2 - 1;
        const _Float16* imb = in + (long)img * Hin * Win * Cin;
        v8f acc = {};
        for (int kt = 0; kt < KT; ++kt) {
            int s = kt >> c2Shift;                       // tap index 0..8
            int ci0 = (kt - (s << c2Shift)) << 5;        // channel base of this k-tile
            int kh = s / 3, kw = s - kh * 3;
            int iy = iy0 + kh, ix = ix0 + kw;
            v8h lo = {}, hi = {};
            if ((unsigned)iy < (unsigned)Hin && (unsigned)ix < (unsigned)Win) {
                const _Float16* p = imb + ((long)iy * Win + ix) * Cin + ci0 + (laneHi << 3);
                lo = *(const v8h*)p;
                hi = *(const v8h*)(p + 16);
            }
            v16h a;
#pragma unroll
            for (int j = 0; j < 8; ++j) { a[j] = lo[j]; a[j + 8] = hi[j]; }
            v16h b = *(const v16h*)(bp + ((long)(nt * KT + kt) * 32 + lane) * 16);
            acc = __builtin_amdgcn_wmma_f32_16x16x32_f16(false, a, false, b, (short)0, acc, false, false);
        }
        int co = nt * 16 + (lane & 15);
        float bv = bias[co];
#pragma unroll
        for (int vv = 0; vv < 8; ++vv) {
            long m2 = mt * 16 + vv + (laneHi << 3);
            int img2 = (int)(m2 / HW); int rem2 = (int)(m2 % HW);
            int oy2 = rem2 / Wout, ox2 = rem2 - oy2 * Wout;
            float val = fmaxf(acc[vv] + bv, 0.f);
            long oidx = (((long)img2 * Hout + oy2) * Wout + ox2) * Cout + co;   // NHWC
            if (out_fp16) ((_Float16*)out)[oidx] = (_Float16)val;
            else          ((float*)out)[oidx] = val;
        }
    }
}

// ------- avgpool 7x7 blocks over NHWC conv2 out -> packed A fragments [T,B,512] --------
__global__ void pool_pack_kernel(const float* __restrict__ y2, _Float16* __restrict__ apack) {
    int e = blockIdx.x * blockDim.x + threadIdx.x;
    if (e >= 1024 * 512) return;
    int f = e & 511; int m = e >> 9;
    int t = m >> 4;  int b = m & 15;
    int bt = b * 64 + t;                       // conv image index (B-major flatten)
    int c = f >> 2;  int ph = (f >> 1) & 1; int pw = f & 1;
    const float* p = y2 + (((long)bt * 14 + ph * 7) * 14 + pw * 7) * 128 + c;
    float s = 0.f;
    for (int i = 0; i < 7; ++i)
        for (int j = 0; j < 7; ++j) s += p[(i * 14 + j) * 128];
    s *= (1.f / 49.f);
    apack[(long)t * 8192 + packA_slot(b, f)] = (_Float16)s;   // M-tile = t (M = t*16+b)
}

// ---------------- XZ = featpack @ wxpack + bias : M=1024 N=2048 K=512 ------------------
__global__ void xz_gemm_kernel(const _Float16* __restrict__ apack, const _Float16* __restrict__ bpack,
                               const float* __restrict__ bias, float* __restrict__ xz) {
    const int NT = 128, KT = 16;
    int lane = threadIdx.x & 31, laneHi = lane >> 4;
    int wave  = blockIdx.x * (blockDim.x >> 5) + (threadIdx.x >> 5);
    int nWave = gridDim.x * (blockDim.x >> 5);
    for (int tile = wave; tile < 64 * NT; tile += nWave) {
        int mt = tile / NT, nt = tile - mt * NT;
        v8f acc = {};
        for (int kt = 0; kt < KT; ++kt) {
            v16h a = *(const v16h*)(apack + ((long)(mt * KT + kt) * 32 + lane) * 16);
            v16h b = *(const v16h*)(bpack + ((long)(nt * KT + kt) * 32 + lane) * 16);
            acc = __builtin_amdgcn_wmma_f32_16x16x32_f16(false, a, false, b, (short)0, acc, false, false);
        }
        int n = nt * 16 + (lane & 15);
        float bv = bias[n];
#pragma unroll
        for (int vv = 0; vv < 8; ++vv)
            xz[(long)(mt * 16 + vv + (laneHi << 3)) * 2048 + n] = acc[vv] + bv;
    }
}

// ---------------- persistent single-WGP peephole LSTM recurrence -----------------------
__global__ __launch_bounds__(1024) void lstm_rec_kernel(
        const float* __restrict__ xz, const _Float16* __restrict__ whpack,
        const float* __restrict__ pi, const float* __restrict__ pf, const float* __restrict__ po,
        _Float16* __restrict__ apack_next, float* __restrict__ dout, int layer) {
    __shared__ __align__(32) float    cS[16 * 512];
    __shared__ __align__(32) _Float16 hpack[16 * 512];
    int tid = threadIdx.x;
    int lane = tid & 31, laneHi = lane >> 4, w = tid >> 5, ln = lane & 15;
    int f = w * 16 + ln;
    float piv = pi[f], pfv = pf[f], pov = po[f];
    for (int i = tid; i < 16 * 512; i += 1024) { cS[i] = 0.f; hpack[i] = (_Float16)0.f; }
    __syncthreads();
    for (int t = 0; t < 64; ++t) {
        v8f ai = {}, af = {}, ag = {}, ao = {};
        for (int kt = 0; kt < 16; ++kt) {
            v16h a = *(const v16h*)(hpack + (kt * 32 + lane) * 16);
            v16h b0 = *(const v16h*)(whpack + (((long)(0 * 32 + w) * 16 + kt) * 32 + lane) * 16);
            v16h b1 = *(const v16h*)(whpack + (((long)(1 * 32 + w) * 16 + kt) * 32 + lane) * 16);
            v16h b2 = *(const v16h*)(whpack + (((long)(2 * 32 + w) * 16 + kt) * 32 + lane) * 16);
            v16h b3 = *(const v16h*)(whpack + (((long)(3 * 32 + w) * 16 + kt) * 32 + lane) * 16);
            if (kt < 15)   // warm next fragments while WMMAs drain (global_prefetch_b8)
                __builtin_prefetch(whpack + (((long)w * 16 + kt + 1) * 32 + lane) * 16, 0, 0);
            ai = __builtin_amdgcn_wmma_f32_16x16x32_f16(false, a, false, b0, (short)0, ai, false, false);
            af = __builtin_amdgcn_wmma_f32_16x16x32_f16(false, a, false, b1, (short)0, af, false, false);
            ag = __builtin_amdgcn_wmma_f32_16x16x32_f16(false, a, false, b2, (short)0, ag, false, false);
            ao = __builtin_amdgcn_wmma_f32_16x16x32_f16(false, a, false, b3, (short)0, ao, false, false);
        }
        __syncthreads();                       // all GEMM reads of hpack done
        const float* xzr = xz + (long)t * 16 * 2048;
#pragma unroll
        for (int vv = 0; vv < 8; ++vv) {
            int b = vv + (laneHi << 3);
            float zi = ai[vv] + xzr[b * 2048 + f];
            float zf = af[vv] + xzr[b * 2048 + 512 + f];
            float zg = ag[vv] + xzr[b * 2048 + 1024 + f];
            float zo = ao[vv] + xzr[b * 2048 + 1536 + f];
            float c  = cS[b * 512 + f];
            float ig = 1.f / (1.f + expf(-(zi + piv * c)));
            float fg = 1.f / (1.f + expf(-(zf + pfv * c)));
            float cn = fg * c + ig * tanhf(zg);
            float og = 1.f / (1.f + expf(-(zo + pov * cn)));
            float hn = og * tanhf(cn);
            cS[b * 512 + f] = cn;
            hpack[packA_slot(b, f)] = (_Float16)hn;
            if (layer < 2) apack_next[(long)t * 8192 + packA_slot(b, f)] = (_Float16)hn;
            else           dout[((long)b * 64 + t) * 512 + f] = hn;
            if (t == 63) {
                long base = 524288 + (long)(2 * layer) * 8192;
                dout[base + b * 512 + f] = hn;
                dout[base + 8192 + b * 512 + f] = cn;
            }
        }
        __syncthreads();                       // hpack/cS updated for next step
    }
}

extern "C" void kernel_launch(void* const* d_in, const int* in_sizes, int n_in,
                              void* d_out, int out_size, void* d_ws, size_t ws_size,
                              hipStream_t stream) {
    const float* x   = (const float*)d_in[0];
    const float* cw0 = (const float*)d_in[1]; const float* cb0 = (const float*)d_in[2];
    const float* cw1 = (const float*)d_in[3]; const float* cb1 = (const float*)d_in[4];
    const float* cw2 = (const float*)d_in[5]; const float* cb2 = (const float*)d_in[6];
    const float* wx[3] = {(const float*)d_in[7],  (const float*)d_in[13], (const float*)d_in[19]};
    const float* wh[3] = {(const float*)d_in[8],  (const float*)d_in[14], (const float*)d_in[20]};
    const float* bi[3] = {(const float*)d_in[9],  (const float*)d_in[15], (const float*)d_in[21]};
    const float* pi[3] = {(const float*)d_in[10], (const float*)d_in[16], (const float*)d_in[22]};
    const float* pf[3] = {(const float*)d_in[11], (const float*)d_in[17], (const float*)d_in[23]};
    const float* po[3] = {(const float*)d_in[12], (const float*)d_in[18], (const float*)d_in[24]};
    float* out = (float*)d_out;

    char* ws = (char*)d_ws;
    size_t off = 0;
    auto alloc = [&](size_t bytes) -> void* {
        off = (off + 255) & ~(size_t)255;
        void* p = ws + off; off += bytes; return p;
    };
    _Float16* a0 = (_Float16*)alloc((size_t)1024 * 56 * 56 * 32 * 2);   // conv0 out f16 NHWC
    _Float16* a1 = (_Float16*)alloc((size_t)1024 * 28 * 28 * 64 * 2);   // conv1 out f16 NHWC
    float*    a2 = (float*)   alloc((size_t)1024 * 14 * 14 * 128 * 4);  // conv2 out f32 NHWC
    _Float16* packA = (_Float16*)alloc((size_t)1024 * 512 * 2);
    _Float16* packB = (_Float16*)alloc((size_t)1024 * 512 * 2);
    float*    xzbuf = (float*)alloc((size_t)1024 * 2048 * 4);
    _Float16 *wxp[3], *whp[3];
    for (int l = 0; l < 3; ++l) {
        wxp[l] = (_Float16*)alloc((size_t)512 * 2048 * 2);
        whp[l] = (_Float16*)alloc((size_t)512 * 2048 * 2);
    }
    _Float16* cwp1 = (_Float16*)alloc((size_t)288 * 64 * 2);
    _Float16* cwp2 = (_Float16*)alloc((size_t)576 * 128 * 2);

    // conv0 direct (fp32 math, f16 NHWC store)
    {
        int total = 1024 * 32 * 56 * 56;
        conv0_kernel<<<(total + 255) / 256, 256, 0, stream>>>(x, cw0, cb0, a0, total);
    }
    // weight repacks (spatial-major im2col K-order)
    pack_convw_kernel<<<(288 * 64 + 255) / 256, 256, 0, stream>>>(cw1, cwp1, 32, 5, 64);
    pack_convw_kernel<<<(576 * 128 + 255) / 256, 256, 0, stream>>>(cw2, cwp2, 64, 6, 128);
    for (int l = 0; l < 3; ++l) {
        pack_w_kernel<<<4096, 256, 0, stream>>>(wx[l], wxp[l]);
        pack_w_kernel<<<4096, 256, 0, stream>>>(wh[l], whp[l]);
    }
    // conv1: 32->64, 56->28 (tiles = 50176*4, 8 waves/block)
    conv_wmma_kernel<<<25088, 256, 0, stream>>>(a0, cwp1, cb1, a1, 32, 0, 56, 56, 64, 28, 28, 1024, 1);
    // conv2: 64->128, 28->14 (tiles = 12544*8)
    conv_wmma_kernel<<<12544, 256, 0, stream>>>(a1, cwp2, cb2, a2, 64, 1, 28, 28, 128, 14, 14, 1024, 0);
    // pool + pack to [T,B,512] A-fragment layout
    pool_pack_kernel<<<2048, 256, 0, stream>>>(a2, packA);

    _Float16* cur = packA; _Float16* nxt = packB;
    for (int l = 0; l < 3; ++l) {
        xz_gemm_kernel<<<1024, 256, 0, stream>>>(cur, wxp[l], bi[l], xzbuf);
        lstm_rec_kernel<<<1, 1024, 0, stream>>>(xzbuf, whp[l], pi[l], pf[l], po[l], nxt, out, l);
        _Float16* tmp = cur; cur = nxt; nxt = tmp;
    }
}
